// ReverseDiffusion_88261577933278
// MI455X (gfx1250) — matrix operand order repaired and checked by reading, not compiled, not run on hardware
//
#include <hip/hip_runtime.h>
#include <hip/hip_bf16.h>
#include <stdint.h>

// ---------------------------------------------------------------------------
// ReverseDiffusion sampling for MI455X (gfx1250, wave32).
//  - async global->LDS streaming (ASYNCcnt) for the histogram pass
//  - ds atomics for the radix-select histogram
//  - v_wmma_f32_16x16x32_f16 computing the top-50 prefix-sum as D = L * P
// ---------------------------------------------------------------------------

typedef __attribute__((ext_vector_type(16))) _Float16 v16h;
typedef __attribute__((ext_vector_type(8)))  float    v8f;

#define NT      256          // threads per block (8 waves)
#define NBINS   4096
#define BSHIFT  20           // key >> 20 -> 4096 bins (sign+exp+3 mantissa bits)
#define TOPK    50
#define TOPP    0.9f
#define MASKID  50256LL
#define CAP     1024         // candidate buffer cap

__device__ __forceinline__ uint32_t sortkey(float f) {
  // monotonic float -> uint mapping (larger float -> larger key)
  uint32_t u = __float_as_uint(f);
  return ((int32_t)u >= 0) ? (u | 0x80000000u) : ~u;
}

// Low 32 bits of a flat pointer into the shared aperture == LDS byte offset.
__device__ __forceinline__ uint32_t lds_byte_off(const void* p) {
  return (uint32_t)(uintptr_t)p;
}

__device__ __forceinline__ void async_ld_b32(uint32_t lds_off, const float* g) {
  asm volatile("global_load_async_to_lds_b32 %0, %1, off"
               :: "v"(lds_off), "v"(g) : "memory");
}
__device__ __forceinline__ void wait_async_le1() {
  asm volatile("s_wait_asynccnt 1" ::: "memory");
}
__device__ __forceinline__ void wait_async_le0() {
  asm volatile("s_wait_asynccnt 0" ::: "memory");
}

__device__ __forceinline__ float row_uniform(int row) {
  uint64_t z = ((uint64_t)(uint32_t)row + 0x9E3779B97F4A7C15ull) * 0xBF58476D1CE4E5B9ull;
  z ^= z >> 30; z *= 0x94D049BB133111EBull; z ^= z >> 31;
  return (float)((uint32_t)(z >> 40)) * (1.0f / 16777216.0f);   // [0,1)
}

__global__ __launch_bounds__(NT)
void rd_sample_kernel(const float* __restrict__ logits,
                      const long long* __restrict__ x_t,
                      float* __restrict__ out,
                      int vocab) {
  __shared__ uint32_t hist[NBINS];      // 16 KB
  __shared__ float    abuf[2][NT];      // async double buffer, 2 KB
  __shared__ float    cval[CAP];        // 4 KB
  __shared__ int      cidx[CAP];        // 4 KB
  __shared__ float    sval[64];
  __shared__ int      sidx[64];
  __shared__ _Float16 pbuf[64];
  __shared__ float    scan[16 * 16];    // WMMA D matrix (column scans)
  __shared__ float    cumb[64];
  __shared__ uint32_t sh_thr;
  __shared__ uint32_t sh_ncand;

  const int row = blockIdx.x;
  const int tid = threadIdx.x;

  // ---- fast path: only masked positions get resampled (block-uniform) ----
  const long long xv = x_t[row];
  if (xv != MASKID) {
    if (tid == 0) out[row] = (float)xv;
    return;
  }

  const float* __restrict__ rowp = logits + (size_t)row * (size_t)vocab;

#pragma unroll 1
  for (int i = tid; i < NBINS; i += NT) hist[i] = 0u;
  if (tid == 0) sh_ncand = 0u;
  __syncthreads();

  // ---- Phase 1: streamed histogram via async global->LDS (double buffered) ----
  const int T = (vocab + NT - 1) / NT;
  if (tid < vocab)
    async_ld_b32(lds_byte_off(&abuf[0][tid]), rowp + tid);
#pragma unroll 1
  for (int t = 0; t < T; ++t) {
    const int nt = t + 1;
    if (nt < T) {
      const int e = nt * NT + tid;
      if (e < vocab)
        async_ld_b32(lds_byte_off(&abuf[nt & 1][tid]), rowp + e);
      wait_async_le1();          // tile t landed; tile t+1 may be in flight
    } else {
      wait_async_le0();
    }
    const int e = t * NT + tid;
    if (e < vocab) {
      const float v = abuf[t & 1][tid];        // own slot: no barrier needed
      atomicAdd(&hist[sortkey(v) >> BSHIFT], 1u);
    }
  }
  __syncthreads();

  // ---- Phase 2: radix-select the bin containing the K-th largest value ----
  if (tid == 0) {
    uint32_t cum = 0;
    int b = NBINS - 1;
#pragma unroll 1
    for (; b > 0; --b) { cum += hist[b]; if (cum >= TOPK) break; }
    sh_thr = ((uint32_t)b) << BSHIFT;           // keys >= this are candidates
  }
  __syncthreads();

  // ---- Phase 3: gather candidates (row is L2-resident now) ----
  const uint32_t thr = sh_thr;
#pragma unroll 1
  for (int e = tid; e < vocab; e += NT) {
    const float v = rowp[e];
    if (sortkey(v) >= thr) {
      const uint32_t slot = atomicAdd(&sh_ncand, 1u);
      if (slot < CAP) { cval[slot] = v; cidx[slot] = e; }
    }
  }
  if (tid < 64) { sval[tid] = -3.0e38f; sidx[tid] = 0; }
  __syncthreads();
  const int n = (sh_ncand < CAP) ? (int)sh_ncand : CAP;

  // ---- Phase 4: exact rank of each candidate -> top-50 sorted descending ----
#pragma unroll 1
  for (int j = tid; j < n; j += NT) {
    const float    vj = cval[j];
    const uint32_t kj = sortkey(vj);
    const int      ij = cidx[j];
    int r = 0;
#pragma unroll 1
    for (int i = 0; i < n; ++i) {
      const uint32_t ki = sortkey(cval[i]);
      if (ki > kj || (ki == kj && cidx[i] < ij)) ++r;   // stable ties: lower idx first
    }
    if (r < 64) { sval[r] = vj; sidx[r] = ij; }
  }
  __syncthreads();

  const int kk = (n < TOPK) ? n : TOPK;

  // softmax numerators of the kept top-k (f16 is plenty for sampling)
  if (tid < 64) {
    float p = 0.0f;
    if (tid < kk) p = __expf(sval[tid] - sval[0]);
    pbuf[tid] = (_Float16)p;
  }
  __syncthreads();

  // ---- Phase 5: prefix sums via WMMA: D = L16 * P  (wave 0, EXEC all-ones) ----
  // P is 16x4 (col c holds probs 16c..16c+15) embedded in the 32x16 B operand;
  // A holds lower-triangular ones in its first 16 K-slices -> D[:,c] = incl. scan.
  if (tid < 32) {
    const int lane = tid;
    const int m_or_n = lane & 15;
    const int half   = lane >> 4;
    v16h A, B;
#pragma unroll
    for (int j = 0; j < 16; ++j) {
      const int vreg = j >> 1;
      const int k = (vreg < 4) ? (2 * vreg + (j & 1) + 8 * half)
                               : (16 + 2 * (vreg - 4) + (j & 1) + 8 * half);
      A[j] = (k < 16 && k <= m_or_n) ? (_Float16)1.0f : (_Float16)0.0f;
    }
#pragma unroll
    for (int j = 0; j < 16; ++j) {
      float bv = 0.0f;
      const int kb = (j < 8) ? (j + 8 * half) : (16 + (j - 8) + 8 * half);
      if (kb < 16 && m_or_n < 4) bv = (float)pbuf[16 * m_or_n + kb];  // K=kb (table order)
      B[j] = (_Float16)bv;
    }
    v8f C = {};
    C = __builtin_amdgcn_wmma_f32_16x16x32_f16(false, A, false, B,
                                               (short)0, C, false, false);
#pragma unroll
    for (int v = 0; v < 8; ++v)
      scan[(v + 8 * half) * 16 + m_or_n] = C[v];   // scan[m*16+n] = D[m][n]
  }
  __syncthreads();

  // ---- Phase 6: stitch columns, top-p cutoff, inverse-CDF sample ----
  if (tid == 0) {
    float off = 0.0f;
#pragma unroll 1
    for (int i = 0; i < kk; ++i) {
      const int col = i >> 4;
      const int m   = i & 15;
      cumb[i] = scan[m * 16 + col] + off;
      if (m == 15) off += scan[15 * 16 + col];
    }
    const float total = cumb[kk - 1];
    // nucleus cutoff: keep prefix while exclusive cumsum <= p * total
    int Kp = 1;
#pragma unroll 1
    for (int i = 1; i < kk; ++i) {
      if (cumb[i - 1] <= TOPP * total) Kp = i + 1; else break;
    }
    const float Pk = cumb[Kp - 1];
    const float r  = row_uniform(row) * Pk;
    int sel = Kp - 1;
#pragma unroll 1
    for (int i = 0; i < Kp; ++i) {
      if (cumb[i] > r) { sel = i; break; }
    }
    out[row] = (float)sidx[sel];
  }
}

extern "C" void kernel_launch(void* const* d_in, const int* in_sizes, int n_in,
                              void* d_out, int out_size, void* d_ws, size_t ws_size,
                              hipStream_t stream) {
  const float*     logits = (const float*)d_in[0];
  const long long* x_t    = (const long long*)d_in[1];
  float*           out    = (float*)d_out;
  (void)d_ws; (void)ws_size; (void)n_in; (void)out_size;

  const int rows  = in_sizes[1];              // B*S = 2048
  const int vocab = in_sizes[0] / rows;       // 50257

  rd_sample_kernel<<<rows, NT, 0, stream>>>(logits, x_t, out, vocab);
}
